// AdaptiveSoftmaxRNN_29497835389222
// MI455X (gfx1250) — compile-verified, matrix operand order checked by reading, not run on hardware
//
#include <hip/hip_runtime.h>
#include <hip/hip_bf16.h>
#include <math.h>

#define SEQ   128
#define BATCH 32
#define NTOK  4096
#define NINP  512
#define NHID  512
#define C0V   20000
#define C1V   50000
#define VV    100000
#define H1V   256
#define H2V   128
#define HEADN 20002
#define NC1   30000
#define NC2   50000

typedef __bf16 bf16;
typedef bf16  bf16x16 __attribute__((ext_vector_type(16)));
typedef float f32x8   __attribute__((ext_vector_type(8)));

// ---------------------------------------------------------------------------
// WMMA fragment helpers (16x16x32 bf16, wave32; layouts per CDNA5 ISA)
// A (16x32, 16-bit): lanes 0-15 rows 0-15 with K-base 0, lanes 16-31 same rows
// K-base 8; VGPR v holds K = base + (v<4?0:16) + (v&3)*2 pairs.
// B (32x16) mirrors A with N in place of M. C: lane 0-15 -> M=v, 16-31 -> M=v+8,
// N = lane%16.
// ---------------------------------------------------------------------------
__device__ __forceinline__ int kpat(int lane, int v) {
    return ((lane & 16) >> 1) + ((v & 4) << 2) + ((v & 3) << 1);
}

__device__ __forceinline__ bf16x16 load_frag(const bf16* __restrict__ base) {
    bf16x16 f;
    const int lane = threadIdx.x & 31;
#pragma unroll
    for (int v = 0; v < 8; ++v) {
        const int k = kpat(lane, v);
        f[2 * v]     = base[k];
        f[2 * v + 1] = base[k + 1];
    }
    return f;
}

__device__ __forceinline__ f32x8 wmma_bf16(bf16x16 a, bf16x16 b, f32x8 c) {
    return __builtin_amdgcn_wmma_f32_16x16x32_bf16(false, a, false, b, (short)0, c,
                                                   false, false);
}

__device__ __forceinline__ float sigmf(float x) { return 1.0f / (1.0f + expf(-x)); }

// ---------------------------------------------------------------------------
// CDNA5 async global->LDS copy (ASYNCcnt path, CDNA5 ISA).
// Low 32 bits of a generic pointer to __shared__ memory are the wave-relative
// LDS byte address.
// ---------------------------------------------------------------------------
__device__ __forceinline__ unsigned lds_off32(const void* p) {
    return (unsigned)(uintptr_t)p;
}

__device__ __forceinline__ void async_copy_b128(unsigned lds_byte,
                                                const void* gaddr) {
    asm volatile("global_load_async_to_lds_b128 %0, %1, off"
                 :
                 : "v"(lds_byte), "v"(gaddr)
                 : "memory");
}

__device__ __forceinline__ void async_wait0() {
    asm volatile("s_wait_asynccnt 0" ::: "memory");
}

// stage a contiguous [32, K] bf16 panel into LDS with async b128 transfers
__device__ __forceinline__ void stage_panel_async(bf16* lds, const bf16* gsrc,
                                                  int elems, int tid) {
    const unsigned lbase = lds_off32(lds);
    for (int i = tid * 16; i < elems; i += 256 * 16)
        async_copy_b128(lbase + (unsigned)i * 2u, gsrc + i);
    async_wait0();
    __syncthreads();
}

// ---------------------------------------------------------------------------
// init: zero sync counters + loss slot, seed bf16 h-state from h0
// ---------------------------------------------------------------------------
__global__ void init_kernel(const float* __restrict__ h0, bf16* __restrict__ hb0,
                            bf16* __restrict__ hb1, unsigned* __restrict__ sync,
                            float* __restrict__ loss_slot) {
    const int tid = blockIdx.x * blockDim.x + threadIdx.x;
    if (tid == 0) { sync[0] = 0u; sync[1] = 0u; *loss_slot = 0.0f; }
    const int stride = gridDim.x * blockDim.x;
    for (int i = tid; i < BATCH * NHID; i += stride) {
        hb0[i] = (bf16)h0[i];
        hb1[i] = (bf16)h0[BATCH * NHID + i];
    }
}

// ---------------------------------------------------------------------------
// f32 -> bf16 conversion (weights)
// ---------------------------------------------------------------------------
__global__ void cvt_kernel(const float* __restrict__ in, bf16* __restrict__ out, int n) {
    int i = blockIdx.x * blockDim.x + threadIdx.x;
    const int stride = gridDim.x * blockDim.x;
    for (; i < n; i += stride) out[i] = (bf16)in[i];
}

// ---------------------------------------------------------------------------
// adaptive input embedding, routed by token range; one block per token
// ---------------------------------------------------------------------------
__global__ void __launch_bounds__(256) emb_kernel(
    const int* __restrict__ toks, const float* __restrict__ head_emb,
    const float* __restrict__ t1_w2, const float* __restrict__ t1_w1,
    const float* __restrict__ t2_w2, const float* __restrict__ t2_w1,
    bf16* __restrict__ out) {
    __shared__ float v[H1V];
    const int tid = threadIdx.x;
    const int tok = toks[blockIdx.x];
    bf16* op = out + (size_t)blockIdx.x * NINP;
    if (tok < C0V) {
        for (int d = tid; d < NINP; d += 256)
            op[d] = (bf16)head_emb[(size_t)tok * NINP + d];
    } else if (tok < C1V) {
        const int r = tok - C0V;
        if (tid < H1V) v[tid] = t1_w2[(size_t)r * H1V + tid];
        __syncthreads();
        for (int d = tid; d < NINP; d += 256) {
            float s = 0.0f;
            for (int k = 0; k < H1V; ++k) s += v[k] * t1_w1[(size_t)k * NINP + d];
            op[d] = (bf16)s;
        }
    } else {
        const int r = tok - C1V;
        if (tid < H2V) v[tid] = t2_w2[(size_t)r * H2V + tid];
        __syncthreads();
        for (int d = tid; d < NINP; d += 256) {
            float s = 0.0f;
            for (int k = 0; k < H2V; ++k) s += v[k] * t2_w1[(size_t)k * NINP + d];
            op[d] = (bf16)s;
        }
    }
}

// ---------------------------------------------------------------------------
// generic WMMA GEMM: C[M,N] = A[M,K] * W[N,K]^T (+bias0+bias1)
// grid = (M/32, N/64), 256 threads (8 waves, 1 tile each)
// ---------------------------------------------------------------------------
__global__ void __launch_bounds__(256) gemm_kernel(
    const bf16* __restrict__ A, const bf16* __restrict__ W,
    const float* __restrict__ bias0, const float* __restrict__ bias1,
    float* __restrict__ Cf, bf16* __restrict__ Cb, int N, int K) {
    __shared__ bf16 As[32 * 512];
    const int tid = threadIdx.x;
    stage_panel_async(As, A + (size_t)blockIdx.x * 32 * K, 32 * K, tid);
    const int wid = tid >> 5, lane = tid & 31, l15 = lane & 15;
    const int mt = wid & 1, nt = wid >> 1;
    const int col = blockIdx.y * 64 + nt * 16 + l15;
    const float b = (bias0 ? bias0[col] : 0.0f) + (bias1 ? bias1[col] : 0.0f);
    f32x8 acc;
#pragma unroll
    for (int e = 0; e < 8; ++e) acc[e] = b;
    const int arow = mt * 16 + l15;
    for (int kb = 0; kb < K; kb += 32) {
        bf16x16 af = load_frag(&As[arow * K + kb]);
        bf16x16 wf = load_frag(&W[(size_t)col * K + kb]);
        acc = wmma_bf16(af, wf, acc);
    }
#pragma unroll
    for (int e = 0; e < 8; ++e) {
        const int m = mt * 16 + e + ((lane & 16) >> 1);
        const size_t idx = (size_t)(blockIdx.x * 32 + m) * N + col;
        if (Cf) Cf[idx] = acc[e];
        if (Cb) Cb[idx] = (bf16)acc[e];
    }
}

// ---------------------------------------------------------------------------
// LSTM recurrence: 8 persistent blocks; block b owns hidden units [b*64,b*64+64)
// (i/f/g/o gate columns b*64 + {0,512,1024,1536}). Double-buffered bf16 h in
// global; grid barrier via agent-scope acquire/release atomics.
// ---------------------------------------------------------------------------
__global__ void __launch_bounds__(256) lstm_rec_kernel(
    const float* __restrict__ Gx,    // [4096, 2048] precomputed x-gates (+biases)
    const bf16* __restrict__ Whh,    // [2048, 512] bf16
    bf16* __restrict__ hA, bf16* __restrict__ hB,  // double-buffered h [32,512]
    const float* __restrict__ c0s,   // [32, 512] layer slice
    bf16* __restrict__ ybf,          // [4096, 512] layer output (bf16)
    float* __restrict__ hN, float* __restrict__ cN,  // final state (into d_out)
    unsigned* __restrict__ sync_cnt) {
    __shared__ float gates[32 * 256];
    __shared__ float cst[32 * 64];
    const int tid = threadIdx.x;
    const int blk = blockIdx.x;  // 0..7
    for (int i = tid; i < 32 * 64; i += 256) {
        const int bat = i >> 6, jl = i & 63;
        cst[i] = c0s[bat * NHID + blk * 64 + jl];
    }
    const int wid = tid >> 5, lane = tid & 31, l15 = lane & 15;
    const int mt = wid & 1, g4 = wid >> 1;  // gate group (i/f/g/o)
    const int arow = mt * 16 + l15;

    for (int t = 0; t < SEQ; ++t) {
        const bf16* hr = (t & 1) ? hB : hA;
        bf16* hw = (t & 1) ? hA : hB;
        for (int nt = 0; nt < 4; ++nt) {
            const int physcol = g4 * 512 + blk * 64 + nt * 16 + l15;
            f32x8 acc;
#pragma unroll
            for (int e = 0; e < 8; ++e) {
                const int m = mt * 16 + e + ((lane & 16) >> 1);
                acc[e] = Gx[(size_t)(t * BATCH + m) * 2048 + physcol];
            }
            for (int kb = 0; kb < NHID; kb += 32) {
                bf16x16 af = load_frag(&hr[arow * NHID + kb]);
                bf16x16 wf = load_frag(&Whh[(size_t)physcol * NHID + kb]);
                acc = wmma_bf16(af, wf, acc);
            }
#pragma unroll
            for (int e = 0; e < 8; ++e) {
                const int m = mt * 16 + e + ((lane & 16) >> 1);
                gates[m * 256 + g4 * 64 + nt * 16 + l15] = acc[e];
            }
        }
        __syncthreads();
        for (int i = tid; i < 32 * 64; i += 256) {
            const int bat = i >> 6, jl = i & 63;
            const float ig = gates[bat * 256 + jl];
            const float fg = gates[bat * 256 + 64 + jl];
            const float gg = gates[bat * 256 + 128 + jl];
            const float og = gates[bat * 256 + 192 + jl];
            const float c = sigmf(fg) * cst[i] + sigmf(ig) * tanhf(gg);
            const float h = sigmf(og) * tanhf(c);
            cst[i] = c;
            const int col = blk * 64 + jl;
            hw[bat * NHID + col] = (bf16)h;
            ybf[(size_t)(t * BATCH + bat) * NHID + col] = (bf16)h;
            if (t == SEQ - 1) {
                hN[bat * NHID + col] = h;
                cN[bat * NHID + col] = c;
            }
        }
        __threadfence();
        __syncthreads();
        if (tid == 0) {
            __hip_atomic_fetch_add(sync_cnt, 1u, __ATOMIC_RELEASE,
                                   __HIP_MEMORY_SCOPE_AGENT);
            const unsigned tgt = 8u * (unsigned)(t + 1);
            while (__hip_atomic_load(sync_cnt, __ATOMIC_ACQUIRE,
                                     __HIP_MEMORY_SCOPE_AGENT) < tgt)
                __builtin_amdgcn_s_sleep(2);
        }
        __syncthreads();
    }
}

// ---------------------------------------------------------------------------
// fused WMMA GEMM + online log-softmax over one cluster; block = 32 rows,
// sweeps all NC columns; captures target logit and up to 2 special columns.
// ---------------------------------------------------------------------------
__global__ void __launch_bounds__(256) softmax_gemm_kernel(
    const bf16* __restrict__ X,     // [4096, K]
    const bf16* __restrict__ W,     // [NC, K]
    const float* __restrict__ bias, // [NC] or null
    const int* __restrict__ targets,
    int NC, int K, int tlow, int thigh, int scol0, int scol1,
    float* __restrict__ out_lp, float* __restrict__ out_s0,
    float* __restrict__ out_s1) {
    __shared__ bf16 Xs[32 * 512];
    __shared__ float tcap[32], scap0[32], scap1[32];
    __shared__ float wmaxs[32][4], wsums[32][4];
    __shared__ int capcol[32];
    const int tid = threadIdx.x;
    if (tid < 32) {
        const int t = targets[blockIdx.x * 32 + tid];
        capcol[tid] = (t >= tlow && t < thigh) ? (t - tlow) : -1;
        tcap[tid] = 0.0f; scap0[tid] = 0.0f; scap1[tid] = 0.0f;
    }
    stage_panel_async(Xs, X + (size_t)blockIdx.x * 32 * K, 32 * K, tid);
    const int wid = tid >> 5, lane = tid & 31, l15 = lane & 15;
    const int mt = wid & 1, nt = wid >> 1;
    const int arow = mt * 16 + l15;
    float rmax[8], rsum[8];
#pragma unroll
    for (int e = 0; e < 8; ++e) { rmax[e] = -3.0e38f; rsum[e] = 0.0f; }

    for (int nb = 0; nb < NC; nb += 64) {
        const int col = nb + nt * 16 + l15;
        const int colc = (col < NC) ? col : (NC - 1);
        f32x8 acc;
#pragma unroll
        for (int e = 0; e < 8; ++e) acc[e] = 0.0f;
        for (int kb = 0; kb < K; kb += 32) {
            bf16x16 af = load_frag(&Xs[arow * K + kb]);
            bf16x16 wf = load_frag(&W[(size_t)colc * K + kb]);
            acc = wmma_bf16(af, wf, acc);
        }
        if (col + 64 < NC)
            __builtin_prefetch(&W[(size_t)(col + 64) * K], 0, 1);
        const float bv = bias ? bias[colc] : 0.0f;
#pragma unroll
        for (int e = 0; e < 8; ++e) {
            float v = acc[e] + bv;
            if (col >= NC) v = -3.0e38f;
            const int m = mt * 16 + e + ((lane & 16) >> 1);
            if (col == capcol[m]) tcap[m] = v;
            if (col == scol0) scap0[m] = v;
            if (col == scol1) scap1[m] = v;
            // per-lane online log-sum-exp over this lane's column subset
            const float nm = fmaxf(rmax[e], v);
            rsum[e] = rsum[e] * expf(rmax[e] - nm) + expf(v - nm);
            rmax[e] = nm;
        }
    }
    // merge the 16 column-lanes of each half-wave (row lives in one 16-lane group)
#pragma unroll
    for (int e = 0; e < 8; ++e) {
        for (int o2 = 1; o2 < 16; o2 <<= 1) {
            const float om = __shfl_xor(rmax[e], o2, 16);
            const float os = __shfl_xor(rsum[e], o2, 16);
            const float nm = fmaxf(rmax[e], om);
            rsum[e] = rsum[e] * expf(rmax[e] - nm) + os * expf(om - nm);
            rmax[e] = nm;
        }
    }
    if (l15 == 0) {
#pragma unroll
        for (int e = 0; e < 8; ++e) {
            const int m = mt * 16 + e + ((lane & 16) >> 1);
            wmaxs[m][nt] = rmax[e];
            wsums[m][nt] = rsum[e];
        }
    }
    __syncthreads();
    if (tid < 32) {
        float fm = -3.0e38f;
        for (int j = 0; j < 4; ++j) fm = fmaxf(fm, wmaxs[tid][j]);
        float fs = 0.0f;
        for (int j = 0; j < 4; ++j) fs += wsums[tid][j] * expf(wmaxs[tid][j] - fm);
        const float denom = fm + logf(fs);
        const int grow = blockIdx.x * 32 + tid;
        out_lp[grow] = tcap[tid] - denom;
        if (out_s0) out_s0[grow] = scap0[tid] - denom;
        if (out_s1) out_s1[grow] = scap1[tid] - denom;
    }
}

// ---------------------------------------------------------------------------
// final select + mean NLL
// ---------------------------------------------------------------------------
__global__ void __launch_bounds__(256) final_kernel(
    const int* __restrict__ targets, const float* __restrict__ lp0,
    const float* __restrict__ lpc0, const float* __restrict__ lpc1,
    const float* __restrict__ lp1, const float* __restrict__ lp2,
    float* __restrict__ out, float* __restrict__ loss_slot) {
    __shared__ float red[256];
    const int n = blockIdx.x * 256 + threadIdx.x;
    const int t = targets[n];
    float v;
    if (t < C0V)      v = lp0[n];
    else if (t < C1V) v = lp1[n] + lpc0[n];
    else              v = lp2[n] + lpc1[n];
    out[n] = v;
    red[threadIdx.x] = v;
    __syncthreads();
    for (int s = 128; s > 0; s >>= 1) {
        if (threadIdx.x < s) red[threadIdx.x] += red[threadIdx.x + s];
        __syncthreads();
    }
    if (threadIdx.x == 0) atomicAdd(loss_slot, -red[0] / (float)NTOK);
}

// ---------------------------------------------------------------------------
extern "C" void kernel_launch(void* const* d_in, const int* in_sizes, int n_in,
                              void* d_out, int out_size, void* d_ws, size_t ws_size,
                              hipStream_t stream) {
    (void)in_sizes; (void)n_in; (void)out_size; (void)ws_size;
    const int*   input_ids = (const int*)d_in[0];
    const float* h0        = (const float*)d_in[1];
    const float* c0        = (const float*)d_in[2];
    const int*   targets   = (const int*)d_in[3];
    const float* head_emb  = (const float*)d_in[4];
    const float* head_w    = (const float*)d_in[5];
    const float* head_b    = (const float*)d_in[6];
    const float* t1_w1     = (const float*)d_in[7];
    const float* t1_w2     = (const float*)d_in[8];
    const float* t2_w1     = (const float*)d_in[9];
    const float* t2_w2     = (const float*)d_in[10];
    const float* w_ih0     = (const float*)d_in[11];
    const float* w_hh0     = (const float*)d_in[12];
    const float* b_ih0     = (const float*)d_in[13];
    const float* b_hh0     = (const float*)d_in[14];
    const float* w_ih1     = (const float*)d_in[15];
    const float* w_hh1     = (const float*)d_in[16];
    const float* b_ih1     = (const float*)d_in[17];
    const float* b_hh1     = (const float*)d_in[18];
    float* out = (float*)d_out;

    char* ws = (char*)d_ws;
    size_t off = 0;
    auto alloc = [&](size_t bytes) -> char* {
        char* p = ws + off;
        off = (off + bytes + 255) & ~(size_t)255;
        return p;
    };
    bf16* head_w_bf = (bf16*)alloc((size_t)HEADN * 512 * 2);
    bf16* t1_w1_bf  = (bf16*)alloc((size_t)H1V * 512 * 2);
    bf16* t1_w2_bf  = (bf16*)alloc((size_t)NC1 * H1V * 2);
    bf16* t2_w1_bf  = (bf16*)alloc((size_t)H2V * 512 * 2);
    bf16* t2_w2_bf  = (bf16*)alloc((size_t)NC2 * H2V * 2);
    bf16* wih0_bf   = (bf16*)alloc((size_t)2048 * 512 * 2);
    bf16* whh0_bf   = (bf16*)alloc((size_t)2048 * 512 * 2);
    bf16* wih1_bf   = (bf16*)alloc((size_t)2048 * 512 * 2);
    bf16* whh1_bf   = (bf16*)alloc((size_t)2048 * 512 * 2);
    bf16* emb_bf    = (bf16*)alloc((size_t)NTOK * 512 * 2);
    bf16* y0_bf     = (bf16*)alloc((size_t)NTOK * 512 * 2);
    bf16* y1_bf     = (bf16*)alloc((size_t)NTOK * 512 * 2);
    bf16* a1_bf     = (bf16*)alloc((size_t)NTOK * H1V * 2);
    bf16* a2_bf     = (bf16*)alloc((size_t)NTOK * H2V * 2);
    float* Gx       = (float*)alloc((size_t)NTOK * 2048 * 4);
    bf16* hb0a      = (bf16*)alloc((size_t)BATCH * NHID * 2);
    bf16* hb0b      = (bf16*)alloc((size_t)BATCH * NHID * 2);
    bf16* hb1a      = (bf16*)alloc((size_t)BATCH * NHID * 2);
    bf16* hb1b      = (bf16*)alloc((size_t)BATCH * NHID * 2);
    float* lp0      = (float*)alloc((size_t)NTOK * 4);
    float* lpc0     = (float*)alloc((size_t)NTOK * 4);
    float* lpc1     = (float*)alloc((size_t)NTOK * 4);
    float* lp1      = (float*)alloc((size_t)NTOK * 4);
    float* lp2      = (float*)alloc((size_t)NTOK * 4);
    unsigned* syncc = (unsigned*)alloc(256);

    // d_out layout: out[4096] | hN[2*32*512] | cN[2*32*512] | loss[1]
    float* hN_base = out + NTOK;
    float* cN_base = out + NTOK + 2 * BATCH * NHID;
    float* loss_slot = out + NTOK + 4 * BATCH * NHID;

    init_kernel<<<32, 256, 0, stream>>>(h0, hb0a, hb1a, syncc, loss_slot);

    auto cvt = [&](const float* src, bf16* dst, size_t n) {
        int blocks = (int)((n + 255) / 256);
        if (blocks > 2048) blocks = 2048;
        cvt_kernel<<<blocks, 256, 0, stream>>>(src, dst, (int)n);
    };
    cvt(head_w, head_w_bf, (size_t)HEADN * 512);
    cvt(t1_w1, t1_w1_bf, (size_t)H1V * 512);
    cvt(t1_w2, t1_w2_bf, (size_t)NC1 * H1V);
    cvt(t2_w1, t2_w1_bf, (size_t)H2V * 512);
    cvt(t2_w2, t2_w2_bf, (size_t)NC2 * H2V);
    cvt(w_ih0, wih0_bf, (size_t)2048 * 512);
    cvt(w_hh0, whh0_bf, (size_t)2048 * 512);
    cvt(w_ih1, wih1_bf, (size_t)2048 * 512);
    cvt(w_hh1, whh1_bf, (size_t)2048 * 512);

    emb_kernel<<<NTOK, 256, 0, stream>>>(input_ids, head_emb, t1_w2, t1_w1,
                                         t2_w2, t2_w1, emb_bf);

    // layer 0
    gemm_kernel<<<dim3(NTOK / 32, 2048 / 64), 256, 0, stream>>>(
        emb_bf, wih0_bf, b_ih0, b_hh0, Gx, nullptr, 2048, 512);
    lstm_rec_kernel<<<8, 256, 0, stream>>>(Gx, whh0_bf, hb0a, hb0b, c0, y0_bf,
                                           hN_base, cN_base, &syncc[0]);
    // layer 1
    gemm_kernel<<<dim3(NTOK / 32, 2048 / 64), 256, 0, stream>>>(
        y0_bf, wih1_bf, b_ih1, b_hh1, Gx, nullptr, 2048, 512);
    lstm_rec_kernel<<<8, 256, 0, stream>>>(Gx, whh1_bf, hb1a, hb1b,
                                           c0 + BATCH * NHID, y1_bf,
                                           hN_base + BATCH * NHID,
                                           cN_base + BATCH * NHID, &syncc[1]);

    // tail projections a1 = x @ t1_w1^T, a2 = x @ t2_w1^T (bf16 out)
    gemm_kernel<<<dim3(NTOK / 32, H1V / 64), 256, 0, stream>>>(
        y1_bf, t1_w1_bf, nullptr, nullptr, nullptr, a1_bf, H1V, 512);
    gemm_kernel<<<dim3(NTOK / 32, H2V / 64), 256, 0, stream>>>(
        y1_bf, t2_w1_bf, nullptr, nullptr, nullptr, a2_bf, H2V, 512);

    // fused online-softmax GEMMs
    softmax_gemm_kernel<<<NTOK / 32, 256, 0, stream>>>(
        y1_bf, head_w_bf, head_b, targets, HEADN, 512, 0, C0V, C0V, C0V + 1,
        lp0, lpc0, lpc1);
    softmax_gemm_kernel<<<NTOK / 32, 256, 0, stream>>>(
        a1_bf, t1_w2_bf, nullptr, targets, NC1, H1V, C0V, C1V, -1, -1,
        lp1, nullptr, nullptr);
    softmax_gemm_kernel<<<NTOK / 32, 256, 0, stream>>>(
        a2_bf, t2_w2_bf, nullptr, targets, NC2, H2V, C1V, VV, -1, -1,
        lp2, nullptr, nullptr);

    final_kernel<<<NTOK / 256, 256, 0, stream>>>(targets, lp0, lpc0, lpc1,
                                                 lp1, lp2, out, loss_slot);
}
